// ConvOnlyNet_83425444758217
// MI455X (gfx1250) — 8-bit matrix operand order repaired and checked by reading, compiled, not run on hardware
//
#include <hip/hip_runtime.h>

typedef int v8i __attribute__((ext_vector_type(8)));

#define BB   16
#define CIN  4
#define COUT 8
#define HH   512
#define WW   512
#define KK   36
#define LDS_PITCH 520

__device__ __forceinline__ unsigned pk4(unsigned a, unsigned b, unsigned c, unsigned d) {
  return (a & 0xFFu) | ((b & 0xFFu) << 8) | ((c & 0xFFu) << 16) | ((d & 0xFFu) << 24);
}

__device__ __forceinline__ unsigned qw(const float* wm, float inv_sw, int k) {
  if (k >= KK) return 0u;
  float v = rintf(wm[k] * inv_sw);
  v = fminf(fmaxf(v, -7.f), 7.f);
  return (unsigned)((int)v) & 0xFFu;
}

__device__ __forceinline__ v8i build_A(const float* w, float s_w, int lane) {
  v8i A = {0, 0, 0, 0, 0, 0, 0, 0};
  const int m = lane & 15;
  if (m < COUT) {
    const float inv_sw = 1.0f / s_w;
    const float* wm = w + m * KK;
    const bool hi = lane >= 16;
    const int k0 = hi ? 16 : 0;
    const int k1 = hi ? 24 : 8;
    A[0] = (int)pk4(qw(wm, inv_sw, k0 + 0), qw(wm, inv_sw, k0 + 1), qw(wm, inv_sw, k0 + 2), qw(wm, inv_sw, k0 + 3));
    A[1] = (int)pk4(qw(wm, inv_sw, k0 + 4), qw(wm, inv_sw, k0 + 5), qw(wm, inv_sw, k0 + 6), qw(wm, inv_sw, k0 + 7));
    A[2] = (int)pk4(qw(wm, inv_sw, k1 + 0), qw(wm, inv_sw, k1 + 1), qw(wm, inv_sw, k1 + 2), qw(wm, inv_sw, k1 + 3));
    A[3] = (int)pk4(qw(wm, inv_sw, k1 + 4), qw(wm, inv_sw, k1 + 5), qw(wm, inv_sw, k1 + 6), qw(wm, inv_sw, k1 + 7));
    A[4] = hi ? 0 : (int)pk4(qw(wm, inv_sw, 32), qw(wm, inv_sw, 33), qw(wm, inv_sw, 34), qw(wm, inv_sw, 35));
  }
  return A;
}

__device__ __forceinline__ void stage(unsigned char* xs, const unsigned char* xq,
                                      int b, int h, int tid) {
  if (tid < 24) {
    const int g = tid >> 1;
    xs[g * LDS_PITCH + ((tid & 1) ? 516 : 3)] = 0;
  }
#pragma unroll
  for (int it = 0; it < 6; ++it) {
    const int idx = it * 256 + tid;
    const int g = idx >> 7;
    const int i = idx & 127;
    const int r = g % 3;
    const int cin = g / 3;
    const int gh = h + r - 1;
    unsigned v = 0;
    if ((unsigned)gh < (unsigned)HH)
      v = ((const unsigned*)(xq + ((((b * CIN + cin) << 9) + gh) << 9)))[i];
    ((unsigned*)(xs + g * LDS_PITCH + 4))[i] = v;
  }
}

__device__ __forceinline__ void build_P(const unsigned char* xs, int p, unsigned P[9]) {
  unsigned D[12];
#pragma unroll
  for (int g = 0; g < 12; ++g)
    __builtin_memcpy(&D[g], xs + (g * LDS_PITCH + 3 + p), 4);
#pragma unroll
  for (int i = 0; i < 9; ++i) {
    const int lo = i + i / 3;
    const int ph = i % 3;
    const unsigned sel = (ph == 0) ? 0x04020100u : ((ph == 1) ? 0x05040201u : 0x06050402u);
    P[i] = __builtin_amdgcn_perm(D[lo + 1], D[lo], sel);
  }
}

__device__ __forceinline__ void conv_slices(v8i A, const unsigned P[9], bool hi, v8i d[4]) {
  const unsigned S0 = hi ? P[4] : P[0];
  const unsigned S1 = hi ? P[5] : P[1];
  const unsigned S2 = hi ? P[6] : P[2];
  const unsigned S3 = hi ? P[7] : P[3];
  const unsigned S4 = hi ? 0u   : P[8];
#pragma unroll
  for (int s = 0; s < 4; ++s) {
    const unsigned sh = 2u * (unsigned)s;
    const unsigned M = 0x03030303u;
    v8i Bv = {0, 0, 0, 0, 0, 0, 0, 0};
    Bv[0] = (int)((S0 >> sh) & M);
    Bv[1] = (int)((S1 >> sh) & M);
    Bv[2] = (int)((S2 >> sh) & M);
    Bv[3] = (int)((S3 >> sh) & M);
    Bv[4] = (int)((S4 >> sh) & M);
    v8i C = {0, 0, 0, 0, 0, 0, 0, 0};
    d[s] = __builtin_amdgcn_wmma_i32_16x16x64_iu8(true, A, false, Bv, C, false, false);
  }
}

__global__ __launch_bounds__(256) void k_quant(const float* __restrict__ x,
                                               const float* __restrict__ s_in_p,
                                               unsigned char* __restrict__ xq,
                                               int n4, int* __restrict__ gmax) {
  const int i = blockIdx.x * 256 + threadIdx.x;
  if (i == 0) *gmax = 0;
  if (i >= n4) return;
  const float inv_si = 1.0f / *s_in_p;
  const float4 v = ((const float4*)x)[i];
  const unsigned b0 = (unsigned)(int)fminf(fmaxf(rintf(v.x * inv_si), 0.f), 255.f);
  const unsigned b1 = (unsigned)(int)fminf(fmaxf(rintf(v.y * inv_si), 0.f), 255.f);
  const unsigned b2 = (unsigned)(int)fminf(fmaxf(rintf(v.z * inv_si), 0.f), 255.f);
  const unsigned b3 = (unsigned)(int)fminf(fmaxf(rintf(v.w * inv_si), 0.f), 255.f);
  ((unsigned*)xq)[i] = pk4(b0, b1, b2, b3);
}

__global__ __launch_bounds__(256) void k_maxpsum(const unsigned char* __restrict__ xq,
                                                 const float* __restrict__ weight,
                                                 const float* __restrict__ s_w_p,
                                                 int* __restrict__ gmax) {
  __shared__ unsigned int xsw[12 * LDS_PITCH / 4];
  __shared__ int smax;
  unsigned char* xs = (unsigned char*)xsw;
  const int tid = threadIdx.x;
  if (tid == 0) smax = 0;
  const int b = blockIdx.x >> 9;
  const int h = blockIdx.x & 511;
  stage(xs, xq, b, h, tid);
  __syncthreads();

  const int lane = tid & 31;
  const int wave = tid >> 5;
  const bool hi = lane >= 16;
  const v8i A = build_A(weight, *s_w_p, lane);

  int mx = 0, mn = 0;
  unsigned P[9];
  build_P(xs, (wave * 4) * 16 + (lane & 15), P);
  v8i d[4];
  conv_slices(A, P, hi, d);
#pragma unroll
  for (int t = 0; t < 4; ++t) {
    unsigned Pn[9];
    if (t < 3) build_P(xs, (wave * 4 + t + 1) * 16 + (lane & 15), Pn);
#pragma unroll
    for (int s = 0; s < 4; ++s) {
#pragma unroll
      for (int j = 0; j < 8; ++j) {
        const int v = d[s][j];
        mx = mx > v ? mx : v;
        mn = mn < v ? mn : v;
      }
    }
    if (t < 3) conv_slices(A, Pn, hi, d);
  }
  const int lmax = mx > -mn ? mx : -mn;
  atomicMax(&smax, lmax);
  __syncthreads();
  if (tid == 0) atomicMax(gmax, smax);
}

__global__ __launch_bounds__(256) void k_conv_out(const unsigned char* __restrict__ xq,
                                                  const float* __restrict__ weight,
                                                  const float* __restrict__ bias,
                                                  const float* __restrict__ s_in_p,
                                                  const float* __restrict__ s_w_p,
                                                  const float* __restrict__ s_out_p,
                                                  const int* __restrict__ gmax,
                                                  float* __restrict__ out) {
  __shared__ unsigned int xsw[12 * LDS_PITCH / 4];
  unsigned char* xs = (unsigned char*)xsw;
  const int tid = threadIdx.x;
  const int b = blockIdx.x >> 9;
  const int h = blockIdx.x & 511;
  stage(xs, xq, b, h, tid);
  __syncthreads();

  const int lane = tid & 31;
  const int wave = tid >> 5;
  const bool hi = lane >= 16;
  const float s_in = *s_in_p, s_w = *s_w_p, s_out = *s_out_p;
  const v8i A = build_A(weight, s_w, lane);

  const float ADC1 = 1.0f / 63.0f;
  const float pm = fmaxf((float)(*gmax), 1e-6f);
  const float gain = fminf(fmaxf(rintf(127.0f / (pm * ADC1)), 1.0f), 255.0f);
  const float step = 1.0f / (gain * ADC1);
  const float inv_step = 1.0f / step;
  const float inv_sout = 1.0f / s_out;
  const float step_scale = step * s_in * s_w;

  float bj[8];
#pragma unroll
  for (int j = 0; j < 8; ++j) bj[j] = bias[j];

  unsigned P[9];
  build_P(xs, (wave * 4) * 16 + (lane & 15), P);
  v8i d[4];
  conv_slices(A, P, hi, d);
#pragma unroll
  for (int t = 0; t < 4; ++t) {
    const int tile = wave * 4 + t;
    unsigned Pn[9];
    if (t < 3) build_P(xs, (tile + 1) * 16 + (lane & 15), Pn);
#pragma unroll
    for (int j = 0; j < 8; ++j) {
      const float q0 = fminf(fmaxf(rintf((float)d[0][j] * inv_step), -127.f), 127.f);
      const float q1 = fminf(fmaxf(rintf((float)d[1][j] * inv_step), -127.f), 127.f);
      const float q2 = fminf(fmaxf(rintf((float)d[2][j] * inv_step), -127.f), 127.f);
      const float q3 = fminf(fmaxf(rintf((float)d[3][j] * inv_step), -127.f), 127.f);
      const float accq = q0 + 4.f * q1 + 16.f * q2 + 64.f * q3;
      const float o = accq * step_scale + bj[j];
      const float q = fminf(fmaxf(rintf(o * inv_sout), -127.f), 127.f) * s_out;
      if (lane < 16)
        out[((((b * COUT + j) << 9) + h) << 9) + tile * 16 + lane] = q;
    }
    if (t < 3) conv_slices(A, Pn, hi, d);
  }
}

extern "C" void kernel_launch(void* const* d_in, const int* in_sizes, int n_in,
                              void* d_out, int out_size, void* d_ws, size_t ws_size,
                              hipStream_t stream) {
  const float* x      = (const float*)d_in[0];
  const float* weight = (const float*)d_in[1];
  const float* bias   = (const float*)d_in[2];
  const float* s_in   = (const float*)d_in[3];
  const float* s_w    = (const float*)d_in[4];
  const float* s_out  = (const float*)d_in[5];
  float* out = (float*)d_out;

  int* gmax = (int*)d_ws;
  unsigned char* xq = (unsigned char*)d_ws + 64;

  const int N = BB * CIN * HH * WW;
  const int n4 = N / 4;
  k_quant<<<(n4 + 255) / 256, 256, 0, stream>>>(x, s_in, xq, n4, gmax);

  const dim3 grid(BB * HH);
  k_maxpsum<<<grid, 256, 0, stream>>>(xq, weight, s_w, gmax);
  k_conv_out<<<grid, 256, 0, stream>>>(xq, weight, bias, s_in, s_w, s_out, gmax, out);
}
